// EinFFT_16063177687677
// MI455X (gfx1250) — matrix operand order repaired and checked by reading, not compiled, not run on hardware
//
#include <hip/hip_runtime.h>

// EinFFT for MI455X (gfx1250): FFT(f32, LDS Stockham) -> complex block-MLP via
// v_wmma_f32_16x16x32_f16 -> inverse FFT. Memory-bound problem (~1.3GB HBM
// traffic over 23.3TB/s); GEMMs (~77 GFLOP) go through the matrix cores.

#define BQ   8
#define NSEQ 4096
#define CCH  768
#define NBLK 4
#define BS   192
#define LAMBDA 0.01f
#define PI_F 3.14159265358979f

typedef __attribute__((ext_vector_type(16))) _Float16 v16h;
typedef __attribute__((ext_vector_type(8)))  float    v8f;

// ---------------------------------------------------------------------------
// A-fragment loader: 16x32 f16 A matrix, contiguous k order (same as B).
// lane L: m = L%16, h = L/16 ; v16h element e -> K = e + 16*h
// => two runs of 8 consecutive halves at kb+16h and kb+16h+8.
static __device__ __forceinline__ v16h load_A_tile(const _Float16* A, int lda,
                                                   int m, int h, int kb) {
  v16h r;
  const _Float16* p0 = A + m * lda + kb + 16 * h;
  const _Float16* p1 = p0 + 8;
#pragma unroll
  for (int e = 0; e < 8; ++e) { r[e] = p0[e]; r[8 + e] = p1[e]; }
  return r;
}

static __device__ __forceinline__ float softshrink(float v) {
  return v > LAMBDA ? v - LAMBDA : (v < -LAMBDA ? v + LAMBDA : 0.f);
}

// ---------------------------------------------------------------------------
// Convert complex weights f32 [part(2)][blk(4)][din(192)][nout(192)] into f16
// transposed [part][blk][nout][din] so B fragments are one 32B load per lane.
__global__ __launch_bounds__(256) void einfft_cvt_w(
    const float* __restrict__ w1, const float* __restrict__ w2,
    _Float16* __restrict__ w1T, _Float16* __restrict__ w2T) {
  long idx = (long)blockIdx.x * 256 + threadIdx.x;
  const long per = 2L * NBLK * BS * BS;  // 294912
  if (idx >= per) return;
  int din  = (int)(idx % BS);
  int nout = (int)((idx / BS) % BS);
  int pb   = (int)(idx / ((long)BS * BS));  // part*4 + blk
  long src = ((long)pb * BS + din) * BS + nout;
  long dst = ((long)pb * BS + nout) * BS + din;
  w1T[dst] = (_Float16)w1[src];
  w2T[dst] = (_Float16)w2[src];
}

// ---------------------------------------------------------------------------
// Stockham radix-2 autosort FFT, length 4096, one column (batch, channel) per
// workgroup, 256 threads, 64KB LDS double buffer. inverse=0: real input ->
// complex out. inverse=1: complex input -> real output (outIm == nullptr).
__global__ __launch_bounds__(256) void einfft_fft_seq(
    const float* __restrict__ inRe, const float* __restrict__ inIm,
    float* __restrict__ outRe, float* __restrict__ outIm, int inverse) {
  __shared__ float l0[NSEQ], l1[NSEQ], l2[NSEQ], l3[NSEQ];
  float* aRe = l0; float* aIm = l1; float* bRe = l2; float* bIm = l3;
  const int tid = threadIdx.x;
  const long base =
      (long)(blockIdx.x / CCH) * NSEQ * CCH + (blockIdx.x % CCH);

  for (int i = tid; i < NSEQ; i += 256) {
    aRe[i] = inRe[base + (long)i * CCH];
    aIm[i] = inIm ? inIm[base + (long)i * CCH] : 0.f;
  }
  __syncthreads();

  const float sgn = inverse ? PI_F : -PI_F;
#pragma unroll 1
  for (int s = 0; s < 12; ++s) {
    const int m = 1 << s;
    const int l = (NSEQ / 2) >> s;
#pragma unroll
    for (int q = 0; q < 8; ++q) {
      const int id = tid + (q << 8);           // 0..2047 butterflies
      const int j = id >> s;                   // j in [0, l)
      const int k = id & (m - 1);              // k in [0, m)
      float sv, cv;
      __sincosf(sgn * (float)j / (float)l, &sv, &cv);
      const int i0 = k + j * m;                // l*m == 2048 invariant
      const float ur = aRe[i0], ui = aIm[i0];
      const float vr = aRe[i0 + 2048], vi = aIm[i0 + 2048];
      const float dr = ur - vr, di = ui - vi;
      const int o = k + 2 * j * m;
      bRe[o] = ur + vr;           bIm[o] = ui + vi;
      bRe[o + m] = dr * cv - di * sv;
      bIm[o + m] = dr * sv + di * cv;
    }
    __syncthreads();
    float* t;
    t = aRe; aRe = bRe; bRe = t;
    t = aIm; aIm = bIm; bIm = t;
  }
  // 12 swaps (even) -> result back in l0/l1 region pointed by aRe/aIm.
  for (int i = tid; i < NSEQ; i += 256) {
    outRe[base + (long)i * CCH] = aRe[i];
    if (outIm) outIm[base + (long)i * CCH] = aIm[i];
  }
}

// ---------------------------------------------------------------------------
// 4-point DFT across the block axis (channels d, d+192, d+384, d+576),
// in-place, with ortho scale 1/sqrt(4096*4) = 1/128 folded in per direction.
__global__ __launch_bounds__(256) void einfft_fft_blk(
    float* __restrict__ Sre, float* __restrict__ Sim, int inverse) {
  long idx = (long)blockIdx.x * 256 + threadIdx.x;
  const long total = (long)BQ * NSEQ * BS;
  if (idx >= total) return;
  long r = idx / BS;
  int d = (int)(idx - r * BS);
  long p = r * CCH + d;
  float x0r = Sre[p],          x0i = Sim[p];
  float x1r = Sre[p + BS],     x1i = Sim[p + BS];
  float x2r = Sre[p + 2 * BS], x2i = Sim[p + 2 * BS];
  float x3r = Sre[p + 3 * BS], x3i = Sim[p + 3 * BS];
  const float sc = 0.0078125f;  // 1/128
  const float sg = inverse ? -1.f : 1.f;
  float e0r = x0r + x2r, e0i = x0i + x2i;
  float e1r = x0r - x2r, e1i = x0i - x2i;
  float o0r = x1r + x3r, o0i = x1i + x3i;
  float o1r = x1r - x3r, o1i = x1i - x3i;
  Sre[p]          = sc * (e0r + o0r);      Sim[p]          = sc * (e0i + o0i);
  Sre[p + BS]     = sc * (e1r + sg * o1i); Sim[p + BS]     = sc * (e1i - sg * o1r);
  Sre[p + 2 * BS] = sc * (e0r - o0r);      Sim[p + 2 * BS] = sc * (e0i - o0i);
  Sre[p + 3 * BS] = sc * (e1r - sg * o1i); Sim[p + 3 * BS] = sc * (e1i + sg * o1r);
}

// ---------------------------------------------------------------------------
// Complex 2-layer block MLP, in-place on the spectrum, via WMMA f16->f32.
// Grid: (32768/16 row tiles, 4 blocks), 128 threads = 4 wave32.
// Each wave computes 3 n-tiles of 16 (4*3*16 = 192 output features).
__global__ __launch_bounds__(128) void einfft_mlp(
    const float* __restrict__ bias1, const float* __restrict__ bias2,
    const _Float16* __restrict__ w1T, const _Float16* __restrict__ w2T,
    float* __restrict__ Sre, float* __restrict__ Sim) {
  __shared__ _Float16 Ar[16 * 200], Ai[16 * 200];
  __shared__ _Float16 R1[16 * 200], I1[16 * 200];

  const int blk = blockIdx.y;
  const long rowBase = (long)blockIdx.x * 16;
  const int tid = threadIdx.x;
  const int lane = tid & 31, wave = tid >> 5;
  const int mlan = lane & 15, h = lane >> 4;

  // Stage 16x192 complex rows to LDS as f16.
  for (int idx = tid; idx < 16 * BS; idx += 128) {
    int m = idx / BS, d = idx - m * BS;
    long g = (rowBase + m) * CCH + (long)blk * BS + d;
    Ar[m * 200 + d] = (_Float16)Sre[g];
    Ai[m * 200 + d] = (_Float16)Sim[g];
  }
  __syncthreads();

  // ---- Layer 1: r1 = relu(Xr*W1r - Xi*W1i + b1r); i1 = relu(Xr*W1i + Xi*W1r + b1i)
#pragma unroll 1
  for (int t = 0; t < 3; ++t) {
    const int n0 = (wave * 3 + t) * 16;
    v8f accR = {}, accI = {};
    const _Float16* wr = w1T + ((long)((0 * NBLK + blk) * BS + n0 + mlan) * BS);
    const _Float16* wi = w1T + ((long)((1 * NBLK + blk) * BS + n0 + mlan) * BS);
#pragma unroll
    for (int kt = 0; kt < 6; ++kt) {
      const int kb = kt * 32;
      v16h aR = load_A_tile(Ar, 200, mlan, h, kb);
      v16h aI = load_A_tile(Ai, 200, mlan, h, kb);
      v16h bR = *(const v16h*)(wr + kb + h * 16);
      v16h bI = *(const v16h*)(wi + kb + h * 16);
      v16h aIn;
#pragma unroll
      for (int e = 0; e < 16; ++e) aIn[e] = -aI[e];
      accR = __builtin_amdgcn_wmma_f32_16x16x32_f16(false, aR,  false, bR, (short)0, accR, false, false);
      accR = __builtin_amdgcn_wmma_f32_16x16x32_f16(false, aIn, false, bI, (short)0, accR, false, false);
      accI = __builtin_amdgcn_wmma_f32_16x16x32_f16(false, aR,  false, bI, (short)0, accI, false, false);
      accI = __builtin_amdgcn_wmma_f32_16x16x32_f16(false, aI,  false, bR, (short)0, accI, false, false);
    }
    const float br = bias1[(0 * NBLK + blk) * BS + n0 + mlan];
    const float bi = bias1[(1 * NBLK + blk) * BS + n0 + mlan];
#pragma unroll
    for (int j = 0; j < 8; ++j) {           // C/D layout: m = j + 8*h, n = mlan
      const int m = j + 8 * h;
      R1[m * 200 + n0 + mlan] = (_Float16)fmaxf(accR[j] + br, 0.f);
      I1[m * 200 + n0 + mlan] = (_Float16)fmaxf(accI[j] + bi, 0.f);
    }
  }
  __syncthreads();

  // ---- Layer 2: r2 = r1*W2r - i1*W2i + b2r ; i2 = r1*W2i + i1*W2r + b2i ; softshrink
#pragma unroll 1
  for (int t = 0; t < 3; ++t) {
    const int n0 = (wave * 3 + t) * 16;
    v8f accR = {}, accI = {};
    const _Float16* wr = w2T + ((long)((0 * NBLK + blk) * BS + n0 + mlan) * BS);
    const _Float16* wi = w2T + ((long)((1 * NBLK + blk) * BS + n0 + mlan) * BS);
#pragma unroll
    for (int kt = 0; kt < 6; ++kt) {
      const int kb = kt * 32;
      v16h aR = load_A_tile(R1, 200, mlan, h, kb);
      v16h aI = load_A_tile(I1, 200, mlan, h, kb);
      v16h bR = *(const v16h*)(wr + kb + h * 16);
      v16h bI = *(const v16h*)(wi + kb + h * 16);
      v16h aIn;
#pragma unroll
      for (int e = 0; e < 16; ++e) aIn[e] = -aI[e];
      accR = __builtin_amdgcn_wmma_f32_16x16x32_f16(false, aR,  false, bR, (short)0, accR, false, false);
      accR = __builtin_amdgcn_wmma_f32_16x16x32_f16(false, aIn, false, bI, (short)0, accR, false, false);
      accI = __builtin_amdgcn_wmma_f32_16x16x32_f16(false, aR,  false, bI, (short)0, accI, false, false);
      accI = __builtin_amdgcn_wmma_f32_16x16x32_f16(false, aI,  false, bR, (short)0, accI, false, false);
    }
    const float br = bias2[(0 * NBLK + blk) * BS + n0 + mlan];
    const float bi = bias2[(1 * NBLK + blk) * BS + n0 + mlan];
#pragma unroll
    for (int j = 0; j < 8; ++j) {
      const int m = j + 8 * h;
      long g = (rowBase + m) * CCH + (long)blk * BS + n0 + mlan;
      Sre[g] = softshrink(accR[j] + br);
      Sim[g] = softshrink(accI[j] + bi);
    }
  }
}

// ---------------------------------------------------------------------------
extern "C" void kernel_launch(void* const* d_in, const int* in_sizes, int n_in,
                              void* d_out, int out_size, void* d_ws,
                              size_t ws_size, hipStream_t stream) {
  const float* x  = (const float*)d_in[0];
  const float* w1 = (const float*)d_in[1];
  const float* b1 = (const float*)d_in[2];
  const float* w2 = (const float*)d_in[3];
  const float* b2 = (const float*)d_in[4];
  float* out = (float*)d_out;

  char* ws = (char*)d_ws;
  const size_t SZ = (size_t)BQ * NSEQ * CCH;        // 25,165,824 elems
  float* Sre = (float*)ws;                          // [B,N,C] f32 spectrum re
  float* Sim = (float*)(ws + SZ * 4);               // [B,N,C] f32 spectrum im
  _Float16* w1T = (_Float16*)(ws + 2 * SZ * 4);     // f16 transposed weights
  _Float16* w2T = (_Float16*)(ws + 2 * SZ * 4 + 2L * NBLK * BS * BS * 2);

  // 1) weights -> f16, transposed to [nout][din]
  einfft_cvt_w<<<(2 * NBLK * BS * BS + 255) / 256, 256, 0, stream>>>(w1, w2, w1T, w2T);
  // 2) forward FFT along seq (real -> complex)
  einfft_fft_seq<<<BQ * CCH, 256, 0, stream>>>(x, nullptr, Sre, Sim, 0);
  // 3) forward 4-pt DFT across blocks (+ ortho 1/128), in place
  const long tot = (long)BQ * NSEQ * BS;
  einfft_fft_blk<<<(int)((tot + 255) / 256), 256, 0, stream>>>(Sre, Sim, 0);
  // 4) complex block MLP via WMMA, in place
  dim3 g(BQ * NSEQ / 16, NBLK);
  einfft_mlp<<<g, 128, 0, stream>>>(b1, b2, w1T, w2T, Sre, Sim);
  // 5) inverse 4-pt DFT across blocks (+ ortho 1/128), in place
  einfft_fft_blk<<<(int)((tot + 255) / 256), 256, 0, stream>>>(Sre, Sim, 1);
  // 6) inverse FFT along seq, real part -> d_out
  einfft_fft_seq<<<BQ * CCH, 256, 0, stream>>>(Sre, Sim, out, nullptr, 1);
}
